// SemanticMemoryGraph_46557445488976
// MI455X (gfx1250) — matrix operand order repaired and checked by reading, not compiled, not run on hardware
//
#include <hip/hip_runtime.h>

#define NODE_DIM 768
#define EDGE_DIM 128
#define K1_EDGE (2 * NODE_DIM + EDGE_DIM)   // 1664
#define K1_NODE (2 * NODE_DIM)              // 1536
#define M_TILE 32
#define WAVE_COLS 48
#define NTILES 3        // n-tiles per wave
#define BLOCK_THREADS 512   // 16 waves x 48 cols = 768

typedef __attribute__((ext_vector_type(16))) __bf16 v16bf;
typedef __attribute__((ext_vector_type(8)))  float  v8f;

union FragBF {
    v16bf bf;
    uint4 q[2];
};

__device__ __forceinline__ unsigned short f2bf(float x) {
    unsigned int u = __float_as_uint(x);
    u += 0x7FFFu + ((u >> 16) & 1u);   // round-to-nearest-even
    return (unsigned short)(u >> 16);
}

__device__ __forceinline__ unsigned int pack2bf(float x0, float x1) {
    return (unsigned int)f2bf(x0) | ((unsigned int)f2bf(x1) << 16);
}

__device__ __forceinline__ v8f wmma_bf16(const FragBF& a, const FragBF& b, v8f c) {
    return __builtin_amdgcn_wmma_f32_16x16x32_bf16(
        /*neg_a=*/false, a.bf, /*neg_b=*/false, b.bf,
        /*c_mod=*/(short)0, c, /*reuse_a=*/false, /*reuse_b=*/false);
}

// ---------------------------------------------------------------------------
// Prep kernels (tiny vs. main GEMMs)
// ---------------------------------------------------------------------------

__global__ __launch_bounds__(256) void cvt_f32_bf16(const float* __restrict__ in,
                                                    unsigned short* __restrict__ out,
                                                    long n) {
    long i = (long)blockIdx.x * 256 + threadIdx.x;
    long stride = (long)gridDim.x * 256;
    for (; i < n; i += stride) out[i] = f2bf(in[i]);
}

// W is [K][N] fp32 row-major; Wt is [N][K] bf16 (K contiguous per output column)
__global__ __launch_bounds__(256) void transpose_f32_bf16(const float* __restrict__ W,
                                                          unsigned short* __restrict__ Wt,
                                                          int K, int N) {
    long total = (long)K * N;
    long i = (long)blockIdx.x * 256 + threadIdx.x;
    long stride = (long)gridDim.x * 256;
    for (; i < total; i += stride) {
        long k = i / N;
        long n = i - k * N;
        Wt[n * (long)K + k] = f2bf(W[i]);
    }
}

__global__ __launch_bounds__(256) void zero_f32(float* __restrict__ p, long n) {
    long i = (long)blockIdx.x * 256 + threadIdx.x;
    long stride = (long)gridDim.x * 256;
    for (; i < n; i += stride) p[i] = 0.f;
}

// ---------------------------------------------------------------------------
// Edge message kernel: 32 edges per block, 16 waves, each wave = 32x48 slab
//   h   = relu([x_row | x_col | rel] @ W1 + b1)    (K = 1664)
//   msg = h @ W2 + b2                              (K = 768)
//   agg[row] += msg   (relaxed f32 atomics)
// ---------------------------------------------------------------------------
__global__ __launch_bounds__(BLOCK_THREADS) void edge_message_kernel(
    const unsigned short* __restrict__ nodesb,  // [N][768] bf16
    const int*            __restrict__ eidx,    // [2][E]
    const int*            __restrict__ etype,   // [E]
    const unsigned short* __restrict__ relb,    // [R][128] bf16
    const unsigned short* __restrict__ w1t,     // [768][1664] bf16 (N-major)
    const float*          __restrict__ b1,      // [768]
    const unsigned short* __restrict__ w2t,     // [768][768] bf16 (N-major)
    const float*          __restrict__ b2,      // [768]
    float*                __restrict__ agg,     // [N][768] f32 (atomic target)
    int E)
{
    extern __shared__ __align__(16) unsigned short smem[];
    unsigned short* sA = smem;                          // 32 x 1664 bf16
    unsigned short* sH = smem + M_TILE * K1_EDGE;       // 32 x 768 bf16
    int* sRow  = (int*)(sH + M_TILE * NODE_DIM);
    int* sCol  = sRow + M_TILE;
    int* sType = sCol + M_TILE;

    const int tid  = threadIdx.x;
    const int lane = tid & 31;
    const int wave = tid >> 5;            // 0..15
    const int m    = lane & 15;
    const int half = lane >> 4;
    const int e0   = blockIdx.x * M_TILE;

    if (tid < M_TILE) {
        int e = e0 + tid;
        if (e < E) {
            sRow[tid]  = eidx[e];
            sCol[tid]  = eidx[E + e];
            sType[tid] = etype[e];
        } else {
            sRow[tid] = -1; sCol[tid] = 0; sType[tid] = 0;
        }
    }
    __syncthreads();

    // Gather 32 x 1664 bf16 input tile into LDS (u32 granularity)
    for (int idx = tid; idx < M_TILE * (K1_EDGE / 2); idx += BLOCK_THREADS) {
        int r = idx / (K1_EDGE / 2);
        int c = idx - r * (K1_EDGE / 2);
        int rsrc = sRow[r] < 0 ? 0 : sRow[r];
        unsigned int val;
        if (c < NODE_DIM / 2) {
            val = *(const unsigned int*)(nodesb + (size_t)rsrc * NODE_DIM + 2 * c);
        } else if (c < NODE_DIM) {
            val = *(const unsigned int*)(nodesb + (size_t)sCol[r] * NODE_DIM + (2 * c - NODE_DIM));
        } else {
            val = *(const unsigned int*)(relb + sType[r] * EDGE_DIM + (2 * c - 2 * NODE_DIM));
        }
        *(unsigned int*)(sA + r * K1_EDGE + 2 * c) = val;
    }
    __syncthreads();

    // ---- Stage 1: 32 x 48 slab per wave, K = 1664 ----
    v8f zero = {0.f, 0.f, 0.f, 0.f, 0.f, 0.f, 0.f, 0.f};
    v8f acc[2][NTILES];
#pragma unroll
    for (int mi = 0; mi < 2; ++mi)
#pragma unroll
        for (int t = 0; t < NTILES; ++t) acc[mi][t] = zero;

    for (int k0 = 0; k0 < K1_EDGE; k0 += 32) {
        // Load both A fragments and all B fragments first (keeps loads in flight)
        FragBF a0, a1, b[NTILES];
        const unsigned short* pa0 = sA + m * K1_EDGE + k0 + half * 8;
        const unsigned short* pa1 = sA + (m + 16) * K1_EDGE + k0 + half * 8;
        a0.q[0] = *(const uint4*)pa0;
        a0.q[1] = *(const uint4*)(pa0 + 16);
        a1.q[0] = *(const uint4*)pa1;
        a1.q[1] = *(const uint4*)(pa1 + 16);
        int kb = k0 + half * 8;
#pragma unroll
        for (int t = 0; t < NTILES; ++t) {
            int n = wave * WAVE_COLS + t * 16 + m;
            const unsigned short* pb = w1t + (size_t)n * K1_EDGE + kb;
            b[t].q[0] = *(const uint4*)pb;
            b[t].q[1] = *(const uint4*)(pb + 16);
        }
#pragma unroll
        for (int t = 0; t < NTILES; ++t) {
            acc[0][t] = wmma_bf16(a0, b[t], acc[0][t]);
            acc[1][t] = wmma_bf16(a1, b[t], acc[1][t]);
        }
    }

    // bias + relu + bf16 -> LDS h-tile
#pragma unroll
    for (int t = 0; t < NTILES; ++t) {
        int n = wave * WAVE_COLS + t * 16 + m;
        float bias = b1[n];
#pragma unroll
        for (int mi = 0; mi < 2; ++mi) {
#pragma unroll
            for (int v = 0; v < 8; ++v) {
                float x = acc[mi][t][v] + bias;
                x = fmaxf(x, 0.f);
                int mr = v + half * 8 + mi * 16;
                sH[mr * NODE_DIM + n] = f2bf(x);
            }
        }
    }
    __syncthreads();

    // ---- Stage 2: msg = h @ W2, K = 768 ----
    v8f acc2[2][NTILES];
#pragma unroll
    for (int mi = 0; mi < 2; ++mi)
#pragma unroll
        for (int t = 0; t < NTILES; ++t) acc2[mi][t] = zero;

    for (int k0 = 0; k0 < NODE_DIM; k0 += 32) {
        FragBF a0, a1, b[NTILES];
        const unsigned short* pa0 = sH + m * NODE_DIM + k0 + half * 8;
        const unsigned short* pa1 = sH + (m + 16) * NODE_DIM + k0 + half * 8;
        a0.q[0] = *(const uint4*)pa0;
        a0.q[1] = *(const uint4*)(pa0 + 16);
        a1.q[0] = *(const uint4*)pa1;
        a1.q[1] = *(const uint4*)(pa1 + 16);
        int kb = k0 + half * 8;
#pragma unroll
        for (int t = 0; t < NTILES; ++t) {
            int n = wave * WAVE_COLS + t * 16 + m;
            const unsigned short* pb = w2t + (size_t)n * NODE_DIM + kb;
            b[t].q[0] = *(const uint4*)pb;
            b[t].q[1] = *(const uint4*)(pb + 16);
        }
#pragma unroll
        for (int t = 0; t < NTILES; ++t) {
            acc2[0][t] = wmma_bf16(a0, b[t], acc2[0][t]);
            acc2[1][t] = wmma_bf16(a1, b[t], acc2[1][t]);
        }
    }

    // bias + scatter-add (relaxed agent-scope f32 atomics)
#pragma unroll
    for (int t = 0; t < NTILES; ++t) {
        int n = wave * WAVE_COLS + t * 16 + m;
        float bias = b2[n];
#pragma unroll
        for (int mi = 0; mi < 2; ++mi) {
#pragma unroll
            for (int v = 0; v < 8; ++v) {
                int mr  = v + half * 8 + mi * 16;
                int dst = sRow[mr];
                if (dst >= 0) {
                    float val = acc2[mi][t][v] + bias;
                    __hip_atomic_fetch_add(&agg[(size_t)dst * NODE_DIM + n], val,
                                           __ATOMIC_RELAXED, __HIP_MEMORY_SCOPE_AGENT);
                }
            }
        }
    }
}

// ---------------------------------------------------------------------------
// Node update kernel: 32 nodes per block, 16 waves, each wave = 32x48 slab
//   u   = relu([x | agg] @ AW1 + ab1)    (K = 1536)
//   out = u @ AW2 + ab2                  (K = 768)
// ---------------------------------------------------------------------------
__global__ __launch_bounds__(BLOCK_THREADS) void node_update_kernel(
    const unsigned short* __restrict__ nodesb,  // [N][768] bf16
    const float*          __restrict__ agg,     // [N][768] f32
    const unsigned short* __restrict__ aw1t,    // [768][1536] bf16
    const float*          __restrict__ ab1,
    const unsigned short* __restrict__ aw2t,    // [768][768] bf16
    const float*          __restrict__ ab2,
    float*                __restrict__ out,     // [N][768] f32
    int N)
{
    extern __shared__ __align__(16) unsigned short smem[];
    unsigned short* sA = smem;                      // 32 x 1536 bf16
    unsigned short* sH = smem + M_TILE * K1_NODE;   // 32 x 768 bf16

    const int tid  = threadIdx.x;
    const int lane = tid & 31;
    const int wave = tid >> 5;
    const int m    = lane & 15;
    const int half = lane >> 4;
    const int n0   = blockIdx.x * M_TILE;

    // Build 32 x 1536 input tile: [nodes_bf16 | cvt(agg f32)]
    for (int idx = tid; idx < M_TILE * (K1_NODE / 2); idx += BLOCK_THREADS) {
        int r = idx / (K1_NODE / 2);
        int c = idx - r * (K1_NODE / 2);
        int node = n0 + r;
        if (node >= N) node = 0;
        unsigned int val;
        if (c < NODE_DIM / 2) {
            val = *(const unsigned int*)(nodesb + (size_t)node * NODE_DIM + 2 * c);
        } else {
            const float* p = agg + (size_t)node * NODE_DIM + (2 * c - NODE_DIM);
            val = pack2bf(p[0], p[1]);
        }
        *(unsigned int*)(sA + r * K1_NODE + 2 * c) = val;
    }
    __syncthreads();

    v8f zero = {0.f, 0.f, 0.f, 0.f, 0.f, 0.f, 0.f, 0.f};
    v8f acc[2][NTILES];
#pragma unroll
    for (int mi = 0; mi < 2; ++mi)
#pragma unroll
        for (int t = 0; t < NTILES; ++t) acc[mi][t] = zero;

    // ---- Stage 1: K = 1536 ----
    for (int k0 = 0; k0 < K1_NODE; k0 += 32) {
        FragBF a0, a1, b[NTILES];
        const unsigned short* pa0 = sA + m * K1_NODE + k0 + half * 8;
        const unsigned short* pa1 = sA + (m + 16) * K1_NODE + k0 + half * 8;
        a0.q[0] = *(const uint4*)pa0;
        a0.q[1] = *(const uint4*)(pa0 + 16);
        a1.q[0] = *(const uint4*)pa1;
        a1.q[1] = *(const uint4*)(pa1 + 16);
        int kb = k0 + half * 8;
#pragma unroll
        for (int t = 0; t < NTILES; ++t) {
            int n = wave * WAVE_COLS + t * 16 + m;
            const unsigned short* pb = aw1t + (size_t)n * K1_NODE + kb;
            b[t].q[0] = *(const uint4*)pb;
            b[t].q[1] = *(const uint4*)(pb + 16);
        }
#pragma unroll
        for (int t = 0; t < NTILES; ++t) {
            acc[0][t] = wmma_bf16(a0, b[t], acc[0][t]);
            acc[1][t] = wmma_bf16(a1, b[t], acc[1][t]);
        }
    }

#pragma unroll
    for (int t = 0; t < NTILES; ++t) {
        int n = wave * WAVE_COLS + t * 16 + m;
        float bias = ab1[n];
#pragma unroll
        for (int mi = 0; mi < 2; ++mi) {
#pragma unroll
            for (int v = 0; v < 8; ++v) {
                float x = acc[mi][t][v] + bias;
                x = fmaxf(x, 0.f);
                int mr = v + half * 8 + mi * 16;
                sH[mr * NODE_DIM + n] = f2bf(x);
            }
        }
    }
    __syncthreads();

    // ---- Stage 2: K = 768 ----
    v8f acc2[2][NTILES];
#pragma unroll
    for (int mi = 0; mi < 2; ++mi)
#pragma unroll
        for (int t = 0; t < NTILES; ++t) acc2[mi][t] = zero;

    for (int k0 = 0; k0 < NODE_DIM; k0 += 32) {
        FragBF a0, a1, b[NTILES];
        const unsigned short* pa0 = sH + m * NODE_DIM + k0 + half * 8;
        const unsigned short* pa1 = sH + (m + 16) * NODE_DIM + k0 + half * 8;
        a0.q[0] = *(const uint4*)pa0;
        a0.q[1] = *(const uint4*)(pa0 + 16);
        a1.q[0] = *(const uint4*)pa1;
        a1.q[1] = *(const uint4*)(pa1 + 16);
        int kb = k0 + half * 8;
#pragma unroll
        for (int t = 0; t < NTILES; ++t) {
            int n = wave * WAVE_COLS + t * 16 + m;
            const unsigned short* pb = aw2t + (size_t)n * NODE_DIM + kb;
            b[t].q[0] = *(const uint4*)pb;
            b[t].q[1] = *(const uint4*)(pb + 16);
        }
#pragma unroll
        for (int t = 0; t < NTILES; ++t) {
            acc2[0][t] = wmma_bf16(a0, b[t], acc2[0][t]);
            acc2[1][t] = wmma_bf16(a1, b[t], acc2[1][t]);
        }
    }

#pragma unroll
    for (int t = 0; t < NTILES; ++t) {
        int n = wave * WAVE_COLS + t * 16 + m;
        float bias = ab2[n];
#pragma unroll
        for (int mi = 0; mi < 2; ++mi) {
#pragma unroll
            for (int v = 0; v < 8; ++v) {
                int mr = v + half * 8 + mi * 16;
                int node = n0 + mr;
                if (node < N) {
                    out[(size_t)node * NODE_DIM + n] = acc2[mi][t][v] + bias;
                }
            }
        }
    }
}

// ---------------------------------------------------------------------------
// Host launch
// ---------------------------------------------------------------------------
extern "C" void kernel_launch(void* const* d_in, const int* in_sizes, int n_in,
                              void* d_out, int out_size, void* d_ws, size_t ws_size,
                              hipStream_t stream) {
    const float* nodes = (const float*)d_in[0];
    const int*   eidx  = (const int*)d_in[1];
    const int*   etype = (const int*)d_in[2];
    const float* rel   = (const float*)d_in[3];
    const float* w1    = (const float*)d_in[4];   // [1664][768]
    const float* b1    = (const float*)d_in[5];
    const float* w2    = (const float*)d_in[6];   // [768][768]
    const float* b2    = (const float*)d_in[7];
    const float* aw1   = (const float*)d_in[8];   // [1536][768]
    const float* ab1   = (const float*)d_in[9];
    const float* aw2   = (const float*)d_in[10];  // [768][768]
    const float* ab2   = (const float*)d_in[11];

    const int N = in_sizes[0] / NODE_DIM;
    const int E = in_sizes[2];
    const int R = in_sizes[3] / EDGE_DIM;

    // Workspace carve-out (256-byte aligned regions)
    char* ws = (char*)d_ws;
    size_t off = 0;
    auto carve = [&](size_t bytes) -> void* {
        void* p = ws + off;
        off = (off + bytes + 255) & ~(size_t)255;
        return p;
    };
    unsigned short* nodesb = (unsigned short*)carve((size_t)N * NODE_DIM * 2);
    unsigned short* relb   = (unsigned short*)carve((size_t)R * EDGE_DIM * 2);
    unsigned short* w1t    = (unsigned short*)carve((size_t)NODE_DIM * K1_EDGE * 2);
    unsigned short* w2t    = (unsigned short*)carve((size_t)NODE_DIM * NODE_DIM * 2);
    unsigned short* aw1t   = (unsigned short*)carve((size_t)NODE_DIM * K1_NODE * 2);
    unsigned short* aw2t   = (unsigned short*)carve((size_t)NODE_DIM * NODE_DIM * 2);
    float*          agg    = (float*)carve((size_t)N * NODE_DIM * 4);
    (void)ws_size;

    // Prep: bf16 conversions, weight transposes, zero accumulator
    cvt_f32_bf16<<<2048, 256, 0, stream>>>(nodes, nodesb, (long)N * NODE_DIM);
    cvt_f32_bf16<<<32, 256, 0, stream>>>(rel, relb, (long)R * EDGE_DIM);
    transpose_f32_bf16<<<1024, 256, 0, stream>>>(w1, w1t, K1_EDGE, NODE_DIM);
    transpose_f32_bf16<<<1024, 256, 0, stream>>>(w2, w2t, NODE_DIM, NODE_DIM);
    transpose_f32_bf16<<<1024, 256, 0, stream>>>(aw1, aw1t, K1_NODE, NODE_DIM);
    transpose_f32_bf16<<<1024, 256, 0, stream>>>(aw2, aw2t, NODE_DIM, NODE_DIM);
    zero_f32<<<2048, 256, 0, stream>>>(agg, (long)N * NODE_DIM);

    // Edge messages + scatter
    size_t smem_edge = (size_t)(M_TILE * K1_EDGE + M_TILE * NODE_DIM) * 2
                     + 3 * M_TILE * sizeof(int);
    edge_message_kernel<<<(E + M_TILE - 1) / M_TILE, BLOCK_THREADS, smem_edge, stream>>>(
        nodesb, eidx, etype, relb, w1t, b1, w2t, b2, agg, E);

    // Node update
    size_t smem_node = (size_t)(M_TILE * K1_NODE + M_TILE * NODE_DIM) * 2;
    node_update_kernel<<<(N + M_TILE - 1) / M_TILE, BLOCK_THREADS, smem_node, stream>>>(
        nodesb, agg, aw1t, ab1, aw2t, ab2, (float*)d_out, N);
}
